// Network_42417097017236
// MI455X (gfx1250) — matrix operand order repaired and checked by reading, not compiled, not run on hardware
//
#include <hip/hip_runtime.h>
#include <hip/hip_bf16.h>
#include <stdint.h>

#define B_SZ   256
#define IN_SZ  2048
#define OUT_SZ 4096
#define N_SZ   8192
#define CH     16
#define NCH    (IN_SZ / CH)
#define WV     4                 // waves per block (K-split of history GEMM)

typedef __attribute__((ext_vector_type(16))) __bf16 v16bf;
typedef __attribute__((ext_vector_type(8)))  float  v8f;

union V16BF { v16bf v; uint4 q[2]; uint32_t u[8]; };

__device__ __forceinline__ uint32_t f2bf(float f) {
  uint32_t u = __builtin_bit_cast(uint32_t, f);
  u += 0x7FFFu + ((u >> 16) & 1u);           // round-to-nearest-even
  return u >> 16;
}
__device__ __forceinline__ float bf2f(uint16_t h) {
  uint32_t u = ((uint32_t)h) << 16;
  return __builtin_bit_cast(float, u);
}

// ---------------------------------------------------------------------------
// Prep: pack the 2048x2048 top-left block of A into the exact per-lane WMMA
// B-operand order so the GEMM loop issues two global_load_b128 per K-block:
//   Apk[ ((kb*IN + c)*2 + half)*8 + v ] = bf16(A[c][2kp]) | bf16(A[c][2kp+1]) << 16
// with kp = kb*16 + half*8 + v   (kb = K-block of 32, c = column, half = lane>>4)
// ---------------------------------------------------------------------------
__global__ void pack_At_kernel(const float* __restrict__ A,
                               uint32_t* __restrict__ Apk) {
  int idx  = blockIdx.x * blockDim.x + threadIdx.x;   // 2^21 dwords
  int v    = idx & 7;
  int half = (idx >> 3) & 1;
  int c    = (idx >> 4) & (IN_SZ - 1);
  int kb   = idx >> 15;
  int kp   = kb * 16 + half * 8 + v;
  const float* row = A + (size_t)c * N_SZ + 2 * kp;
  Apk[idx] = f2bf(row[0]) | (f2bf(row[1]) << 16);
}

// ---------------------------------------------------------------------------
// Recurrence: one block (4 waves) per 16-row batch tile; the 16x2048 bf16
// state slab lives in LDS for the whole kernel.  Per 16-column chunk:
//   - history GEMM pre += S(16 x i0) * A^T(i0 x 16) with
//     v_wmma_f32_16x16x32_bf16, K-blocks strided across the 4 waves
//   - partial accumulators reduced through LDS
//   - wave 0 runs the 16-step f32 triangular sweep with tanh + broadcasts
// ---------------------------------------------------------------------------
__global__ void __launch_bounds__(32 * WV)
recurrence_kernel(const float* __restrict__ x, const float* __restrict__ A,
                  const uint32_t* __restrict__ Apk,
                  const float* __restrict__ na, const float* __restrict__ nb,
                  const float* __restrict__ nc, uint16_t* __restrict__ Sg) {
  __shared__ uint16_t S[CH * IN_SZ];            // 64 KB state slab
  __shared__ float Pred[WV - 1][16][16];        // 3 KB partial tiles

  const int tid  = threadIdx.x;
  const int wid  = tid >> 5;                    // wave id 0..3
  const int lane = tid & 31;
  const int half = lane >> 4;                   // 0/1
  const int n    = lane & 15;                   // tile column / A-operand M
  const int rowbase = blockIdx.x * CH;

  {   // zero-init state (faithful: fs0 = 0)
    uint32_t* S32 = (uint32_t*)S;
    for (int i = tid; i < CH * IN_SZ / 2; i += 32 * WV) S32[i] = 0u;
  }
  __syncthreads();

  for (int ci = 0; ci < NCH; ++ci) {
    const int i0 = ci * CH;
    const int nkb = (i0 + 31) >> 5;             // K-blocks of 32 (tail cols are 0)

    // ---- history GEMM, K split across waves ----
    v8f cacc = {};
    for (int kb = wid; kb < nkb; kb += WV) {
      V16BF a, b;
      // A-operand (state rows) from LDS: M = lane&15,
      // lanes<16: K 0..15 ; lanes>=16: K 16..31
      const uint16_t* pa = S + n * IN_SZ + kb * 32 + half * 16;
      a.q[0] = *(const uint4*)(pa);
      a.q[1] = *(const uint4*)(pa + 8);
      // B-operand: 8 contiguous dwords per lane -> two b128 loads
      const uint32_t* pb =
          Apk + (((((size_t)kb * IN_SZ) + (i0 + n)) * 2 + half) << 3);
      b.q[0] = *(const uint4*)(pb);
      b.q[1] = *(const uint4*)(pb + 4);
      // speculative prefetch of this lane's next K-block
      __builtin_prefetch(pb + (size_t)WV * IN_SZ * 16, 0, 1);
      cacc = __builtin_amdgcn_wmma_f32_16x16x32_bf16(
          false, a.v, false, b.v, (short)0, cacc, false, false);
    }

    // ---- reduce partial accumulators through LDS ----
    if (wid != 0) {
#pragma unroll
      for (int v = 0; v < 8; ++v) Pred[wid - 1][half * 8 + v][n] = cacc[v];
    }
    __syncthreads();

    if (wid == 0) {
      // pre = sum of partials + x[:, i0+n]  (C layout: row = half*8+v, col = n)
      float pre[8];
#pragma unroll
      for (int v = 0; v < 8; ++v) {
        const int r = half * 8 + v;
        pre[v] = cacc[v] + Pred[0][r][n] + Pred[1][r][n] + Pred[2][r][n] +
                 x[(size_t)(rowbase + r) * IN_SZ + (i0 + n)];
      }

      // per-lane step params for column i0+n, and old state (read faithfully)
      const float pa_ = na[i0 + n], pb_ = nb[i0 + n], pc_ = nc[i0 + n];
      float st[8];
#pragma unroll
      for (int v = 0; v < 8; ++v)
        st[v] = bf2f(S[(half * 8 + v) * IN_SZ + i0 + n]);

      // lane n preloads A[i0+n][i0 .. i0+15] for the triangular part
      float ar[16];
      {
        const float4* ap = (const float4*)(A + (size_t)(i0 + n) * N_SZ + i0);
        float4 t0 = ap[0], t1 = ap[1], t2 = ap[2], t3 = ap[3];
        ar[0]=t0.x;  ar[1]=t0.y;  ar[2]=t0.z;  ar[3]=t0.w;
        ar[4]=t1.x;  ar[5]=t1.y;  ar[6]=t1.z;  ar[7]=t1.w;
        ar[8]=t2.x;  ar[9]=t2.y;  ar[10]=t2.z; ar[11]=t2.w;
        ar[12]=t3.x; ar[13]=t3.y; ar[14]=t3.z; ar[15]=t3.w;
      }

      // sequential sweep over the 16 chunk columns
#pragma unroll
      for (int k = 0; k < CH; ++k) {
#pragma unroll
        for (int v = 0; v < 8; ++v) {
          float t = tanhf(pa_ * pre[v] + pb_ * st[v] + pc_);
          pre[v] = (n == k) ? t : pre[v];     // finalize column k
        }
        const int src = (lane & 16) | k;      // broadcast within 16-lane half
#pragma unroll
        for (int v = 0; v < 8; ++v) {
          float sk = __shfl(pre[v], src, 32);
          if (n > k) pre[v] += sk * ar[k];    // later cols += A[n,k]*Snew_k
        }
      }

      // commit chunk columns to LDS state as bf16
#pragma unroll
      for (int v = 0; v < 8; ++v)
        S[(half * 8 + v) * IN_SZ + i0 + n] = (uint16_t)f2bf(pre[v]);
    }
    __syncthreads();   // state visible to all waves; Pred safe to overwrite
  }

  // persist the state tile (rows rowbase..rowbase+15) for the output stage
  {
    uint32_t* S32 = (uint32_t*)S;
    uint32_t* G32 = (uint32_t*)(Sg + (size_t)rowbase * IN_SZ);
    for (int i = tid; i < CH * IN_SZ / 2; i += 32 * WV) G32[i] = S32[i];
  }
}

// ---------------------------------------------------------------------------
// Output: out[b] = sum_{j=OUT..N-1} fs[b,j] * w[j-OUT].
// fs columns >= IN were never written (== 0); the generic guard below is
// constant-folded since OUT >= IN at compile time.
// ---------------------------------------------------------------------------
__global__ void out_kernel(const uint16_t* __restrict__ Sg,
                           const float* __restrict__ w,
                           float* __restrict__ out) {
  int b = blockIdx.x * blockDim.x + threadIdx.x;
  if (b >= B_SZ) return;
  float sum = 0.0f;
  for (int j = OUT_SZ; j < N_SZ; ++j) {
    float v = (j < IN_SZ) ? bf2f(Sg[(size_t)b * IN_SZ + j]) : 0.0f;
    sum += v * w[j - OUT_SZ];
  }
  out[b] = sum;
}

extern "C" void kernel_launch(void* const* d_in, const int* in_sizes, int n_in,
                              void* d_out, int out_size, void* d_ws, size_t ws_size,
                              hipStream_t stream) {
  const float* x  = (const float*)d_in[0];
  const float* A  = (const float*)d_in[1];
  const float* w  = (const float*)d_in[2];
  const float* na = (const float*)d_in[3];
  const float* nb = (const float*)d_in[4];
  const float* nc = (const float*)d_in[5];

  uint32_t* Apk = (uint32_t*)d_ws;                                    // 8 MB
  uint16_t* Sg  = (uint16_t*)((char*)d_ws +
                              (size_t)(IN_SZ / 2) * IN_SZ * sizeof(uint32_t)); // 1 MB

  pack_At_kernel<<<(IN_SZ / 2) * IN_SZ / 256, 256, 0, stream>>>(A, Apk);
  recurrence_kernel<<<B_SZ / CH, 32 * WV, 0, stream>>>(x, A, Apk, na, nb, nc, Sg);
  out_kernel<<<1, 256, 0, stream>>>(Sg, w, (float*)d_out);
}
